// MINT_63909113364906
// MI455X (gfx1250) — matrix operand order repaired and checked by reading, not compiled, not run on hardware
//
#include <hip/hip_runtime.h>

typedef __attribute__((ext_vector_type(16))) _Float16 v16h;
typedef __attribute__((ext_vector_type(8)))  float    v8f;

#define NIMG   64
#define KP     32
#define HH     96
#define WW     96
#define HW     (HH*WW)      // 9216
#define CCH    3
#define HIN    128
#define WIN    128
#define NPLANE (NIMG*CCH)   // 192
#define SFS    4

// ---------------------------------------------------------------------------
// Resize weight matrix R[96][128] (antialiased bilinear, row-normalized), f16
// ---------------------------------------------------------------------------
__global__ void k_weights(_Float16* __restrict__ R) {
    int o = blockIdx.x * blockDim.x + threadIdx.x;
    if (o >= HH) return;
    const float scale = (float)HH / (float)HIN;           // 0.75
    float xin = (o + 0.5f) / scale - 0.5f;
    float s = 0.f;
    for (int i = 0; i < HIN; ++i) {
        float t = 1.0f - fabsf((float)i - xin) * scale;
        s += (t > 0.f ? t : 0.f);
    }
    float inv = 1.0f / s;
    for (int i = 0; i < HIN; ++i) {
        float t = 1.0f - fabsf((float)i - xin) * scale;
        t = (t > 0.f ? t : 0.f) * inv;
        R[o * HIN + i] = (_Float16)t;
    }
}

// f32 -> f16 convert
__global__ void k_cvt(const float* __restrict__ src, _Float16* __restrict__ dst, int n) {
    int i = blockIdx.x * blockDim.x + threadIdx.x;
    if (i < n) dst[i] = (_Float16)src[i];
}

// ---------------------------------------------------------------------------
// WMMA helper: one wave computes a 16x16 f32 tile of A(16xK) * Bt(16xK)^T
// A row-major [lda] halves, Bt row-major (N-major) [ldb] halves.  K % 32 == 0.
// Lane layout per CDNA5 ISA (16-bit A 16x32, B 32x16).
// ---------------------------------------------------------------------------
union V16HU { v16h v; float4 f4[2]; };

__device__ inline v8f wmma_tile(const _Float16* __restrict__ arow0,
                                const _Float16* __restrict__ brow0,
                                int K) {
    v8f acc = {};
    for (int k = 0; k < K; k += 32) {
        V16HU a, b;
        // A halves 0..7  : K = k + hs*8 + 0..7   (arow0 already offset by hs*8)
        // A halves 8..15 : K = k + 16 + hs*8 + 0..7
        a.f4[0] = *(const float4*)(arow0 + k);
        a.f4[1] = *(const float4*)(arow0 + k + 16);
        // B halves 0..7 : K = k + hs*8 + 0..7, halves 8..15 : K = k + 16 + hs*8 + 0..7 (brow0 already offset by hs*8)
        b.f4[0] = *(const float4*)(brow0 + k);
        b.f4[1] = *(const float4*)(brow0 + k + 16);
        acc = __builtin_amdgcn_wmma_f32_16x16x32_f16(
            false, a.v, false, b.v, (short)0, acc, false, false);
    }
    return acc;
}

// Pass 1: tmpT[plane][x][i] = sum_j img[plane][i][j] * R[x][j]
// grid = NPLANE * (128/16) * (96/16) = 192*48, block = 32 (one wave)
__global__ void k_wmma_rows(const _Float16* __restrict__ imgh,
                            const _Float16* __restrict__ R,
                            _Float16* __restrict__ tmpT) {
    int t = blockIdx.x;
    int plane = t / 48;
    int r = t % 48;
    int m0 = (r / 6) * 16;      // row tile of img (i)
    int n0 = (r % 6) * 16;      // col tile of output (x)
    int lane = threadIdx.x;
    int hs = lane >> 4;
    int ml = lane & 15;

    const _Float16* A  = imgh + (size_t)plane * HIN * WIN;
    const _Float16* ar = A + (size_t)(m0 + ml) * WIN + hs * 8;
    const _Float16* br = R + (size_t)(n0 + ml) * WIN + hs * 8;
    v8f acc = wmma_tile(ar, br, WIN);

    _Float16* out = tmpT + (size_t)plane * HH * HIN;   // [x][i], 96x128
    int x = n0 + ml;
    for (int rr = 0; rr < 8; ++rr) {
        int i = m0 + rr + hs * 8;
        out[(size_t)x * HIN + i] = (_Float16)acc[rr];
    }
}

// Pass 2: resized[plane][y][x] = sum_i R[y][i] * tmpT[plane][x][i]
// grid = NPLANE * (96/16) * (96/16) = 192*36, block = 32
__global__ void k_wmma_cols(const _Float16* __restrict__ R,
                            const _Float16* __restrict__ tmpT,
                            float* __restrict__ resized) {
    int t = blockIdx.x;
    int plane = t / 36;
    int r = t % 36;
    int m0 = (r / 6) * 16;      // y tile
    int n0 = (r % 6) * 16;      // x tile
    int lane = threadIdx.x;
    int hs = lane >> 4;
    int ml = lane & 15;

    const _Float16* Bt = tmpT + (size_t)plane * HH * HIN;
    const _Float16* ar = R  + (size_t)(m0 + ml) * HIN + hs * 8;
    const _Float16* br = Bt + (size_t)(n0 + ml) * HIN + hs * 8;
    v8f acc = wmma_tile(ar, br, HIN);

    float* out = resized + (size_t)plane * HW;
    int x = n0 + ml;
    for (int rr = 0; rr < 8; ++rr) {
        int y = m0 + rr + hs * 8;
        out[(size_t)y * WW + x] = acc[rr];
    }
}

// ---------------------------------------------------------------------------
// Local entropy: 16-bin KDE, 3x3 zero-padded box mean, channel-averaged.
// one thread per (n, y, x); grid = 64*9216/256 = 2304
// ---------------------------------------------------------------------------
__global__ void k_entropy(const float* __restrict__ resized,
                          float* __restrict__ ent) {
    int id = blockIdx.x * blockDim.x + threadIdx.x;
    if (id >= NIMG * HW) return;
    int n  = id / HW;
    int px = id % HW;
    int y = px / WW, x = px % WW;

    float esum = 0.f;
    for (int c = 0; c < CCH; ++c) {
        const float* img = resized + (size_t)(n * CCH + c) * HW;
        float nv[9]; int nc = 0;
        for (int dy = -1; dy <= 1; ++dy) {
            int yy = y + dy;
            if (yy < 0 || yy >= HH) continue;
            for (int dx = -1; dx <= 1; ++dx) {
                int xx = x + dx;
                if (xx < 0 || xx >= WW) continue;
                nv[nc++] = img[yy * WW + xx];
            }
        }
        float p[16], norm = 0.f;
        for (int j = 0; j < 16; ++j) {
            float bj = (float)j * (1.0f / 15.0f);
            float s = 0.f;
            for (int t = 0; t < nc; ++t) {
                float d = (nv[t] - bj) * 10.0f;     // / BANDWIDTH(0.1)
                s += __expf(-0.5f * d * d);
            }
            p[j] = s * (1.0f / 9.0f);               // zero-pad SAME -> always /9
            norm += p[j];
        }
        float inv = 1.0f / (norm + 1e-10f);
        float e = 0.f;
        for (int j = 0; j < 16; ++j) {
            float ph = p[j] * inv;
            e -= ph * __logf(ph + 1e-10f);
        }
        esum += e;
    }
    ent[id] = esum * (1.0f / (float)CCH);
}

// ---------------------------------------------------------------------------
// Block reduce helpers (blockDim.x == 256)
// ---------------------------------------------------------------------------
__device__ inline float bsum256(float v, float* sm) {
    int t = threadIdx.x;
    sm[t] = v; __syncthreads();
    for (int o = 128; o > 0; o >>= 1) { if (t < o) sm[t] += sm[t + o]; __syncthreads(); }
    float r = sm[0]; __syncthreads(); return r;
}
__device__ inline float bmax256(float v, float* sm) {
    int t = threadIdx.x;
    sm[t] = v; __syncthreads();
    for (int o = 128; o > 0; o >>= 1) { if (t < o) sm[t] = fmaxf(sm[t], sm[t + o]); __syncthreads(); }
    float r = sm[0]; __syncthreads(); return r;
}

__device__ inline float heat(float g) {
    return fminf(fmaxf((g - 0.1f) * 3.5f, 0.f), 1.f);
}

// Per-n aggregation: agg_mask, max(sum gauss), heatmap_size (n==0,kp==0)
// grid = 64, block = 256
__global__ void k_agg(const float* __restrict__ gauss,
                      const float* __restrict__ status,
                      float* __restrict__ mask,
                      float* __restrict__ maxg,
                      float* __restrict__ hmsize) {
    __shared__ float sm[256];
    int n = blockIdx.x;
    const float* g = gauss + (size_t)n * KP * HW;
    float st[KP];
    for (int k = 0; k < KP; ++k) st[k] = status[n * KP + k];
    float lmax = -1e30f, lhm0 = 0.f;
    for (int px = threadIdx.x; px < HW; px += 256) {
        float msum = 0.f, gsum = 0.f;
        for (int k = 0; k < KP; ++k) {
            float gv = g[(size_t)k * HW + px];
            float hm = heat(gv);
            msum += st[k] * hm;
            gsum += gv;
            if (n == 0 && k == 0) lhm0 += hm;
        }
        mask[(size_t)n * HW + px] = fminf(fmaxf(msum, 0.f), 1.f);
        lmax = fmaxf(lmax, gsum);
    }
    float m = bmax256(lmax, sm);
    if (threadIdx.x == 0) maxg[n] = m;
    if (n == 0) {
        float h = bsum256(lhm0, sm);
        if (threadIdx.x == 0) hmsize[0] = h;
    }
}

// Per-n entropy sums. grid = 64, block = 256
__global__ void k_esum(const float* __restrict__ ent,
                       const float* __restrict__ mask,
                       float* __restrict__ re_sum, float* __restrict__ ce_sum,
                       float* __restrict__ mel_num, float* __restrict__ mcel_num) {
    __shared__ float sm[256];
    int n = blockIdx.x;
    int b = n / SFS, s = n % SFS;
    int np = b * SFS + ((s + SFS - 1) % SFS);
    float rs = 0.f, cs = 0.f, mn = 0.f, cn = 0.f;
    for (int px = threadIdx.x; px < HW; px += 256) {
        float re = ent[(size_t)n * HW + px];
        float se = ent[(size_t)np * HW + px];
        float ce = fmaxf(re, se) - se;
        float mk = mask[(size_t)n * HW + px];
        rs += re; cs += ce;
        mn += re * (1.f - mk);
        cn += ce * (1.f - mk);
    }
    float r;
    r = bsum256(rs, sm); if (threadIdx.x == 0) re_sum[n] = r;
    r = bsum256(cs, sm); if (threadIdx.x == 0) ce_sum[n] = r;
    r = bsum256(mn, sm); if (threadIdx.x == 0) mel_num[n] = r;
    r = bsum256(cn, sm); if (threadIdx.x == 0) mcel_num[n] = r;
}

// MI sum per (n, kp):  mi = min(recon, re). grid = 64*32, block = 256
__global__ void k_mi(const float* __restrict__ gauss,
                     const float* __restrict__ status,
                     const float* __restrict__ ent,
                     float* __restrict__ mi_sum) {
    __shared__ float sm[256];
    int n  = blockIdx.x >> 5;
    int kp = blockIdx.x & 31;
    int b = n / SFS, s = n % SFS;
    int np = b * SFS + ((s + SFS - 1) % SFS);
    float stc = status[n * KP + kp];
    float stp = status[np * KP + kp];
    const float* gc = gauss + ((size_t)n * KP + kp) * HW;
    const float* gp = gauss + ((size_t)np * KP + kp) * HW;
    float acc = 0.f;
    for (int px = threadIdx.x; px < HW; px += 256) {
        float a  = stc * heat(gc[px]);
        float sh = stp * heat(gp[px]);
        float re = ent[(size_t)n * HW + px];
        float se = ent[(size_t)np * HW + px];
        float ce = fmaxf(re, se) - se;
        float recon = se * (1.f - a) * (1.f - sh) + re * a + ce * (1.f - a);
        acc += fminf(recon, re);
    }
    float r = bsum256(acc, sm);
    if (threadIdx.x == 0) mi_sum[blockIdx.x] = r;
}

// Final combine: 1 block, 64 threads (one per n)
__global__ void k_final(const float* __restrict__ coords,
                        const float* __restrict__ status,
                        const float* __restrict__ re_sum,
                        const float* __restrict__ ce_sum,
                        const float* __restrict__ mel_num,
                        const float* __restrict__ mcel_num,
                        const float* __restrict__ maxg,
                        const float* __restrict__ hmsize,
                        const float* __restrict__ mi_sum,
                        float* __restrict__ out) {
    __shared__ float sm[64];
    int n = threadIdx.x;                  // 0..63
    int b = n / SFS, s = n % SFS;
    int np = b * SFS + ((s + SFS - 1) % SFS);
    float move = 0.f, stl = 0.f;
    for (int k = 0; k < KP; ++k) {
        float sc = status[n * KP + k];
        float sp = status[np * KP + k];
        float dx = coords[(n * KP + k) * 2 + 0] - coords[(np * KP + k) * 2 + 0];
        float dy = coords[(n * KP + k) * 2 + 1] - coords[(np * KP + k) * 2 + 1];
        move += sqrtf(dx * dx + dy * dy) * sc * sp;
        stl += sc;
    }
    stl *= (1.0f / (float)KP);
    float hs = hmsize[0];
    float rs = re_sum[n];
    float rc = 0.f;
    for (int k = 0; k < KP; ++k) rc += (rs - mi_sum[n * KP + k]);
    rc = rc / ((float)KP * hs);
    float mel  = mel_num[n] / (rs + 1e-10f);
    float mcel = (s == 0) ? 0.f : mcel_num[n] / (ce_sum[n] + 1e-10f);
    float overlap = fmaxf(maxg[n] - 1.5f, 0.f) / (float)KP;
    float itl = rc + 0.1f * move;
    float mint = 100.f * mel + 100.f * mcel + 1.0f * itl
               + 10.f * overlap + (1.f - mel) * 2.5f * stl;
    sm[n] = mint; __syncthreads();
    for (int o = 32; o > 0; o >>= 1) { if (n < o) sm[n] += sm[n + o]; __syncthreads(); }
    if (n == 0) out[0] = sm[0] * (1.0f / (float)NIMG);
}

// ---------------------------------------------------------------------------
extern "C" void kernel_launch(void* const* d_in, const int* in_sizes, int n_in,
                              void* d_out, int out_size, void* d_ws, size_t ws_size,
                              hipStream_t stream) {
    const float* coords = (const float*)d_in[0];   // (64,32,2)
    const float* gauss  = (const float*)d_in[1];   // (64,32,96,96)
    const float* status = (const float*)d_in[2];   // (64,32)
    const float* images = (const float*)d_in[3];   // (64,3,128,128)
    float* out = (float*)d_out;

    char* ws = (char*)d_ws;
    size_t off = 0;
    auto take = [&](size_t bytes) -> char* {
        char* p = ws + off;
        off = (off + bytes + 255) & ~(size_t)255;
        return p;
    };
    _Float16* R       = (_Float16*)take((size_t)HH * HIN * 2);
    _Float16* imgh    = (_Float16*)take((size_t)NPLANE * HIN * WIN * 2);
    _Float16* tmpT    = (_Float16*)take((size_t)NPLANE * HH * HIN * 2);
    float*    resized = (float*)   take((size_t)NPLANE * HW * 4);
    float*    ent     = (float*)   take((size_t)NIMG * HW * 4);
    float*    mask    = (float*)   take((size_t)NIMG * HW * 4);
    float*    re_sum  = (float*)   take(NIMG * 4);
    float*    ce_sum  = (float*)   take(NIMG * 4);
    float*    mel_num = (float*)   take(NIMG * 4);
    float*    mcel_num= (float*)   take(NIMG * 4);
    float*    maxg    = (float*)   take(NIMG * 4);
    float*    hmsize  = (float*)   take(64 * 4);
    float*    mi_sum  = (float*)   take((size_t)NIMG * KP * 4);

    // 1) resize weights + f16 convert
    k_weights<<<3, 32, 0, stream>>>(R);
    int ncvt = NPLANE * HIN * WIN;
    k_cvt<<<(ncvt + 255) / 256, 256, 0, stream>>>(images, imgh, ncvt);

    // 2) resize via WMMA GEMMs
    k_wmma_rows<<<NPLANE * 48, 32, 0, stream>>>(imgh, R, tmpT);
    k_wmma_cols<<<NPLANE * 36, 32, 0, stream>>>(R, tmpT, resized);

    // 3) local entropy
    k_entropy<<<(NIMG * HW) / 256, 256, 0, stream>>>(resized, ent);

    // 4) heatmap aggregation, 5) entropy sums, 6) MI, 7) combine
    k_agg<<<NIMG, 256, 0, stream>>>(gauss, status, mask, maxg, hmsize);
    k_esum<<<NIMG, 256, 0, stream>>>(ent, mask, re_sum, ce_sum, mel_num, mcel_num);
    k_mi<<<NIMG * KP, 256, 0, stream>>>(gauss, status, ent, mi_sum);
    k_final<<<1, 64, 0, stream>>>(coords, status, re_sum, ce_sum, mel_num,
                                  mcel_num, maxg, hmsize, mi_sum, out);
}
